// MultiheadSelfAttention_17910013624825
// MI455X (gfx1250) — matrix operand order repaired and checked by reading, not compiled, not run on hardware
//
#include <hip/hip_runtime.h>
#include <stdint.h>

#define HIDDEN 1024
#define NHEADS 16
#define DH 64
#define BB 4
#define TT 2048

typedef __attribute__((ext_vector_type(16))) __bf16 v16bf;
typedef __attribute__((ext_vector_type(8)))  float  v8f;

union Frag { v16bf v; uint32_t u[8]; };

__device__ __forceinline__ v8f wmma_bf16(v16bf a, v16bf b, v8f c) {
  return __builtin_amdgcn_wmma_f32_16x16x32_bf16(false, a, false, b, (short)0, c,
                                                 false, false);
}

// K index held in A-fragment VGPR v for lane-half h (16-bit A 16x32 layout, ISA 7.12.2)
__device__ __forceinline__ int akk(int v, int h) {
  return 16 * (v >> 2) + 8 * h + 2 * (v & 3);
}

// Async DMA: 16B per lane, global -> LDS, tracked by ASYNCcnt (ISA 10, async ops)
__device__ __forceinline__ void async_b128(uint32_t lds_off, const void* gaddr) {
  asm volatile("global_load_async_to_lds_b128 %0, %1, off"
               :: "v"(lds_off), "v"(gaddr) : "memory");
}
__device__ __forceinline__ void wait_async4() {
  asm volatile("s_wait_asynccnt 0x4" ::: "memory");
}
__device__ __forceinline__ void wait_async0() {
  asm volatile("s_wait_asynccnt 0x0" ::: "memory");
}
__device__ __forceinline__ uint32_t lds_addr(const void* p) {
  return (uint32_t)(uintptr_t)p;  // low 32 bits of generic LDS ptr = LDS byte offset
}

// 16-lane XOR butterfly: quad_perm DPP for xor1/2 (pure VALU), ds_swizzle for xor4/8
__device__ __forceinline__ float dpp_xor1(float x) {
  int i = __float_as_int(x);
  return __int_as_float(__builtin_amdgcn_update_dpp(i, i, 0xB1, 0xF, 0xF, false));
}
__device__ __forceinline__ float dpp_xor2(float x) {
  int i = __float_as_int(x);
  return __int_as_float(__builtin_amdgcn_update_dpp(i, i, 0x4E, 0xF, 0xF, false));
}
__device__ __forceinline__ float swz_xor4(float x) {
  return __int_as_float(__builtin_amdgcn_ds_swizzle(__float_as_int(x), 0x101F));
}
__device__ __forceinline__ float swz_xor8(float x) {
  return __int_as_float(__builtin_amdgcn_ds_swizzle(__float_as_int(x), 0x201F));
}
__device__ __forceinline__ float red16_max(float x) {
  x = fmaxf(x, dpp_xor1(x)); x = fmaxf(x, dpp_xor2(x));
  x = fmaxf(x, swz_xor4(x)); x = fmaxf(x, swz_xor8(x));
  return x;
}
__device__ __forceinline__ float red16_sum(float x) {
  x += dpp_xor1(x); x += dpp_xor2(x);
  x += swz_xor4(x); x += swz_xor8(x);
  return x;
}

#define MSENT (-3.0e38f)  // finite row-max sentinel: keeps exp() guard-free

// ---------------------------------------------------------------------------
// f32 -> bf16 elementwise convert (vectorized)
// ---------------------------------------------------------------------------
__global__ void k_convert(const float* __restrict__ in, __bf16* __restrict__ out,
                          int n) {
  int i = (blockIdx.x * 256 + threadIdx.x) * 4;
  if (i >= n) return;
  float4 f = *(const float4*)(in + i);
  out[i + 0] = (__bf16)f.x; out[i + 1] = (__bf16)f.y;
  out[i + 2] = (__bf16)f.z; out[i + 3] = (__bf16)f.w;
}

// Tiled transpose + convert:  Wt[n*K + k] = W[k*N + n]  (coalesced both ways)
__global__ __launch_bounds__(256) void k_transpose_convert(
    const float* __restrict__ W, __bf16* __restrict__ Wt, int K, int N) {
  __shared__ float T[32][33];
  const int n0 = blockIdx.x * 32, k0 = blockIdx.y * 32;
  const int c = threadIdx.x & 31, r = threadIdx.x >> 5;  // r: 0..7
#pragma unroll
  for (int i = 0; i < 4; i++) {
    int row = r + i * 8;
    T[row][c] = W[(size_t)(k0 + row) * N + n0 + c];
  }
  __syncthreads();
#pragma unroll
  for (int i = 0; i < 4; i++) {
    int row = r + i * 8;
    Wt[(size_t)(n0 + row) * K + k0 + c] = (__bf16)T[c][row];
  }
}

// ---------------------------------------------------------------------------
// WMMA GEMM: C[M,N] = A[M,K] (bf16) * Bt[N,K]^T (bf16) + bias
// 128x128x32 tile, 256 threads = 8 waves (4x2), wave tile 32x64.
// Double-buffered async-to-LDS staging.
// ---------------------------------------------------------------------------
template <bool OUT_BF16, bool DO_ELU>
__global__ __launch_bounds__(256) void k_gemm(
    const __bf16* __restrict__ A, const __bf16* __restrict__ Bt,
    const float* __restrict__ bias, __bf16* __restrict__ outb,
    float* __restrict__ outf, int M, int N, int K, int qcols, float qscale) {
  __shared__ __bf16 As[2][128 * 40];  // pitch 40 (80B, 16B-aligned rows, conflict-free)
  __shared__ __bf16 Bs[2][128 * 40];  // column-major B (rows = n, cols = kk)

  const int tid = threadIdx.x;
  const int lane = tid & 31, wid = tid >> 5;
  const int hl = lane >> 4, ln = lane & 15;
  const int n0 = blockIdx.x * 128, m0 = blockIdx.y * 128;
  const int waveM = (wid & 3) * 32, waveN = (wid >> 2) * 64;

  // 128x32 bf16 tile = 512 x 16B chunks; 2 per thread (4 chunks per row)
  int crow[2], ccol[2];
#pragma unroll
  for (int i = 0; i < 2; i++) {
    int c = tid + 256 * i;
    crow[i] = c >> 2;
    ccol[i] = (c & 3) * 8;
  }

  auto prefetch = [&](int buf, int k0) {
#pragma unroll
    for (int i = 0; i < 2; i++)
      async_b128(lds_addr(&As[buf][crow[i] * 40 + ccol[i]]),
                 A + (size_t)(m0 + crow[i]) * K + k0 + ccol[i]);
#pragma unroll
    for (int i = 0; i < 2; i++)
      async_b128(lds_addr(&Bs[buf][crow[i] * 40 + ccol[i]]),
                 Bt + (size_t)(n0 + crow[i]) * K + k0 + ccol[i]);
  };

  v8f acc[2][4];
#pragma unroll
  for (int i = 0; i < 2; i++)
#pragma unroll
    for (int j = 0; j < 4; j++) acc[i][j] = {};

  prefetch(0, 0);
  const int S = K >> 5;
  for (int kt = 0; kt < S; kt++) {
    const int cur = kt & 1;
    const bool more = (kt + 1 < S);
    if (more) prefetch(cur ^ 1, (kt + 1) * 32);   // overlap DMA with this step's math
    if (more) wait_async4(); else wait_async0();  // my cur-buffer fills have landed
    __syncthreads();                              // everyone's fills have landed

    Frag a[2];
#pragma unroll
    for (int mt = 0; mt < 2; mt++) {
      int m = waveM + mt * 16 + ln;
#pragma unroll
      for (int v = 0; v < 8; v++)
        a[mt].u[v] = *(const uint32_t*)&As[cur][m * 40 + akk(v, hl)];
    }
#pragma unroll
    for (int nt = 0; nt < 4; nt++) {
      Frag bfr;
      int n = waveN + nt * 16 + ln;
#pragma unroll
      for (int v = 0; v < 8; v++)
        bfr.u[v] = *(const uint32_t*)&Bs[cur][n * 40 + akk(v, hl)];
      acc[0][nt] = wmma_bf16(a[0].v, bfr.v, acc[0][nt]);
      acc[1][nt] = wmma_bf16(a[1].v, bfr.v, acc[1][nt]);
    }
    __syncthreads();  // reads of cur closed -> next iter may DMA into it
  }

  // Epilogue: bias, optional column-range scale (Q * 1/sqrt(dh)), optional ELU
#pragma unroll
  for (int nt = 0; nt < 4; nt++) {
    int n = n0 + waveN + nt * 16 + ln;
    float bv = bias[n];
    float s = (n < qcols) ? qscale : 1.0f;
#pragma unroll
    for (int mt = 0; mt < 2; mt++) {
#pragma unroll
      for (int r = 0; r < 8; r++) {
        int m = m0 + waveM + mt * 16 + r + 8 * hl;
        float v = (acc[mt][nt][r] + bv) * s;
        if (DO_ELU) v = v > 0.f ? v : (__expf(v) - 1.f);
        if (OUT_BF16) outb[(size_t)m * N + n] = (__bf16)v;
        else          outf[(size_t)m * N + n] = v;
      }
    }
  }
}

// ---------------------------------------------------------------------------
// Flash attention: one block = 64 query rows of one (b,h); 4 waves x 16 rows.
// qkv layout: [b][t][3*HIDDEN] bf16; Q pre-scaled by 1/8.
// K and V tiles double-buffered via async-to-LDS, both row-major [key][d].
// ---------------------------------------------------------------------------
__global__ __launch_bounds__(128) void k_attn(const __bf16* __restrict__ qkv,
                                              const int* __restrict__ masks,
                                              __bf16* __restrict__ ctx) {
  __shared__ __bf16 Ks[2][32 * 72];   // pitch 72 (144B rows, 16B-aligned)
  __shared__ __bf16 Vs[2][32 * 72];
  __shared__ __bf16 Ps[4][16 * 40];   // per-wave P staging [q][key]
  __shared__ float  Mb[TT];           // additive key-mask bias: 0 or -inf

  const int tid = threadIdx.x;
  const int lane = tid & 31, wid = tid >> 5;
  const int hl = lane >> 4, ln = lane & 15;
  const int bh = blockIdx.y;
  const int b = bh >> 4, h = bh & 15;
  const int q0 = blockIdx.x * 64 + wid * 16;
  const size_t rowstride = 3 * HIDDEN;
  const size_t bt0 = (size_t)b * TT;

  // 32x64 bf16 tile = 256 x 16B chunks; 2 per thread (8 chunks per row)
  int crow[2], ccol[2];
#pragma unroll
  for (int i = 0; i < 2; i++) {
    int c = tid + 128 * i;
    crow[i] = c >> 3;
    ccol[i] = (c & 7) * 8;
  }
  auto prefetchKV = [&](int buf, int kb) {
#pragma unroll
    for (int i = 0; i < 2; i++)
      async_b128(lds_addr(&Ks[buf][crow[i] * 72 + ccol[i]]),
                 qkv + (bt0 + kb * 32 + crow[i]) * rowstride + HIDDEN +
                     (size_t)h * DH + ccol[i]);
#pragma unroll
    for (int i = 0; i < 2; i++)
      async_b128(lds_addr(&Vs[buf][crow[i] * 72 + ccol[i]]),
                 qkv + (bt0 + kb * 32 + crow[i]) * rowstride + 2 * HIDDEN +
                     (size_t)h * DH + ccol[i]);
  };

  prefetchKV(0, 0);

  // Key-mask bias table (coalesced, once per block)
  for (int i = tid; i < TT; i += 128)
    Mb[i] = masks[bt0 + i] ? 0.f : -INFINITY;

  // Q fragments (dh = 64 -> two 16x32 A fragments), loaded once
  Frag qa[2];
  {
    const size_t base = (bt0 + q0 + ln) * rowstride + (size_t)h * DH;
#pragma unroll
    for (int f = 0; f < 2; f++)
#pragma unroll
      for (int v = 0; v < 8; v++)
        qa[f].u[v] = *(const uint32_t*)(qkv + base + 32 * f + akk(v, hl));
  }

  v8f cacc[4];
#pragma unroll
  for (int i = 0; i < 4; i++) cacc[i] = {};
  float mrow[8], lrow[8];
#pragma unroll
  for (int r = 0; r < 8; r++) { mrow[r] = MSENT; lrow[r] = 0.f; }

  for (int kb = 0; kb < TT / 32; kb++) {
    const int cur = kb & 1;
    const bool more = (kb + 1 < TT / 32);
    if (more) prefetchKV(cur ^ 1, kb + 1);
    if (more) wait_async4(); else wait_async0();
    __syncthreads();  // tiles landed (and Mb ready on first iteration)

    const float bias0 = Mb[kb * 32 + ln];
    const float bias1 = Mb[kb * 32 + 16 + ln];

    // S = Q @ K^T : two 16x16 f32 accumulators (keys 0-15 / 16-31)
    v8f s0 = {}, s1 = {};
#pragma unroll
    for (int f = 0; f < 2; f++) {
      Frag bk0, bk1;
#pragma unroll
      for (int v = 0; v < 8; v++) {
        int d = 32 * f + akk(v, hl);
        bk0.u[v] = *(const uint32_t*)&Ks[cur][ln * 72 + d];
        bk1.u[v] = *(const uint32_t*)&Ks[cur][(16 + ln) * 72 + d];
      }
      s0 = wmma_bf16(qa[f].v, bk0.v, s0);
      s1 = wmma_bf16(qa[f].v, bk1.v, s1);
    }

    // Online softmax (guard-free: masked scores are -inf, exp(-inf)=0;
    // finite sentinel keeps mnew finite so no NaN paths)
    float p0[8], p1[8];
#pragma unroll
    for (int r = 0; r < 8; r++) {
      float a0 = s0[r] + bias0;
      float a1 = s1[r] + bias1;
      float mx = red16_max(fmaxf(a0, a1));
      float mnew = fmaxf(mrow[r], mx);
      float e0 = __expf(a0 - mnew);
      float e1 = __expf(a1 - mnew);
      float rs = red16_sum(e0 + e1);
      float sc = __expf(mrow[r] - mnew);
      lrow[r] = lrow[r] * sc + rs;
      mrow[r] = mnew;
      p0[r] = e0; p1[r] = e1;
#pragma unroll
      for (int nt = 0; nt < 4; nt++) cacc[nt][r] *= sc;
    }

    // Stage P (C layout) -> LDS -> reload as A fragment (per-wave private)
#pragma unroll
    for (int r = 0; r < 8; r++) {
      Ps[wid][(r + 8 * hl) * 40 + ln]      = (__bf16)p0[r];
      Ps[wid][(r + 8 * hl) * 40 + 16 + ln] = (__bf16)p1[r];
    }
    asm volatile("s_wait_dscnt 0x0" ::: "memory");  // per-wave LDS RAW fence
    Frag pa;
#pragma unroll
    for (int v = 0; v < 8; v++)
      pa.u[v] = *(const uint32_t*)&Ps[wid][ln * 40 + akk(v, hl)];

    // ctx(16x64) += P(16x32) @ V(32x64): B fragments gathered from row-major V
    const uint16_t* Vr = (const uint16_t*)&Vs[cur][0];
#pragma unroll
    for (int nt = 0; nt < 4; nt++) {
      Frag bv;
#pragma unroll
      for (int v = 0; v < 8; v++) {
        int kk = akk(v, hl), n = nt * 16 + ln;
        uint32_t lo = Vr[kk * 72 + n];
        uint32_t hi = Vr[(kk + 1) * 72 + n];
        bv.u[v] = lo | (hi << 16);
      }
      cacc[nt] = wmma_bf16(pa.v, bv.v, cacc[nt]);
    }
    __syncthreads();  // reads of cur closed -> next iter may DMA into it
  }

  // Normalize, apply query mask, store bf16 ctx [b][t][h*64+d]
#pragma unroll
  for (int r = 0; r < 8; r++) {
    int q = q0 + r + 8 * hl;
    float qm = (float)masks[bt0 + q];
    float inv = lrow[r] > 0.f ? qm / lrow[r] : 0.f;
#pragma unroll
    for (int nt = 0; nt < 4; nt++) {
      int d = nt * 16 + ln;
      ctx[(bt0 + q) * HIDDEN + (size_t)h * DH + d] = (__bf16)(cacc[nt][r] * inv);
    }
  }
}

// ---------------------------------------------------------------------------
extern "C" void kernel_launch(void* const* d_in, const int* in_sizes, int n_in,
                              void* d_out, int out_size, void* d_ws,
                              size_t ws_size, hipStream_t stream) {
  (void)in_sizes; (void)n_in; (void)out_size; (void)ws_size;
  const float* x     = (const float*)d_in[0];
  const int*   masks = (const int*)d_in[1];
  const float* w_qkv = (const float*)d_in[2];
  const float* b_qkv = (const float*)d_in[3];
  const float* w_out = (const float*)d_in[4];
  const float* b_out = (const float*)d_in[5];
  float* out = (float*)d_out;

  char* ws = (char*)d_ws;
  __bf16* Wqkv_t = (__bf16*)(ws);              //  6,291,456 B
  __bf16* Wout_t = (__bf16*)(ws + 6291456);    //  2,097,152 B
  __bf16* xbf    = (__bf16*)(ws + 8388608);    // 16,777,216 B
  __bf16* qkv    = (__bf16*)(ws + 25165824);   // 50,331,648 B
  __bf16* ctxb   = (__bf16*)(ws + 75497472);   // 16,777,216 B

  // Precision staging: weights f32 [K,N] -> bf16 [N,K]; x f32 -> bf16
  k_transpose_convert<<<dim3(3072 / 32, 1024 / 32), 256, 0, stream>>>(
      w_qkv, Wqkv_t, 1024, 3072);
  k_transpose_convert<<<dim3(1024 / 32, 1024 / 32), 256, 0, stream>>>(
      w_out, Wout_t, 1024, 1024);
  k_convert<<<dim3(8192 * 1024 / 4 / 256), 256, 0, stream>>>(x, xbf, 8192 * 1024);

  // QKV projection -> bf16 qkv, Q columns (<1024) scaled by 1/sqrt(64)
  k_gemm<true, false><<<dim3(3072 / 128, 8192 / 128), 256, 0, stream>>>(
      xbf, Wqkv_t, b_qkv, qkv, nullptr, 8192, 3072, 1024, 1024, 0.125f);

  // Masked flash attention -> bf16 ctx
  k_attn<<<dim3(TT / 64, BB * NHEADS), 128, 0, stream>>>(qkv, masks, ctxb);

  // Output projection + bias + ELU -> f32 out
  k_gemm<false, true><<<dim3(1024 / 128, 8192 / 128), 256, 0, stream>>>(
      ctxb, Wout_t, b_out, nullptr, out, 8192, 1024, 1024, 0, 1.0f);
}
